// VSSBlock_78494822301830
// MI455X (gfx1250) — matrix operand order repaired and checked by reading, not compiled, not run on hardware
//
#include <hip/hip_runtime.h>
#include <hip/hip_bf16.h>

// ---------------------------------------------------------------------------
// VSSBlock (2-D Mamba) for gfx1250 (MI455X, wave32, WMMA + TDM).
// - All GEMMs: v_wmma_f32_16x16x32_bf16, bf16 fragments loaded directly
//   (weights pre-converted to bf16 once; activations stored bf16 where they
//   are only consumed as WMMA A-operands). Pipeline is HBM-bandwidth bound
//   (~1 GB traffic @ 23.3 TB/s), so bf16 storage is the main lever.
// - Selective scan: TENSOR_LOAD_TO_LDS double-buffered staging of B_t/C_t,
//   synchronized with s_wait_tensorcnt + workgroup barriers.
// ---------------------------------------------------------------------------

#define B_SZ   16
#define HID    96
#define DI     192
#define DI2    384
#define DSTATE 16
#define DTRANK 12
#define LPIX   4096            // 64*64
#define NBL    (B_SZ * LPIX)   // 65536 rows (M) for all GEMMs

typedef __attribute__((ext_vector_type(16))) __bf16 v16bf;
typedef __attribute__((ext_vector_type(8)))  __bf16 v8bf;
typedef __attribute__((ext_vector_type(8)))  float  v8f;
typedef __attribute__((ext_vector_type(4)))  unsigned int u32x4;
typedef __attribute__((ext_vector_type(8)))  int i32x8;
typedef __attribute__((ext_vector_type(4)))  int i32x4;

__device__ __forceinline__ float silu_f(float v) {
    return v / (1.f + __expf(-v));
}

// ---------------------------------------------------------------------------
// TDM: 1-D contiguous f32 strip (nelem floats) from global -> LDS.
// D# per CDNA5 ISA §8.3/§8.4: group0 {count=1, lds_addr, global_addr, type=2};
// group1 {data_size=4B, tensor_dim0=nelem, tensor_dim1=1, tile_dim0=nelem,
// tile_dim1=1, dim0_stride=nelem}. Groups 2/3 zero (<=2D). Tracked: TENSORcnt.
// ---------------------------------------------------------------------------
__device__ __forceinline__ void tdm_load_f32(unsigned lds_addr,
                                             const float* gptr,
                                             unsigned nelem) {
    const unsigned long long ga = (unsigned long long)(uintptr_t)gptr;
    u32x4 g0;
    g0[0] = 1u;                                               // count=1 (valid)
    g0[1] = lds_addr;                                         // bits[63:32]
    g0[2] = (unsigned)(ga & 0xFFFFFFFFu);                     // global_addr lo
    g0[3] = (unsigned)((ga >> 32) & 0x1FFFFFFu) | (2u << 30); // addr hi | type=2
    i32x8 g1;
    g1[0] = (int)(2u << 16);                                  // data_size = 4B
    g1[1] = (int)((nelem & 0xFFFFu) << 16);                   // tensor_dim0 lo16
    g1[2] = (int)((nelem >> 16) & 0xFFFFu) | (1 << 16);       // td0 hi | td1=1
    g1[3] = (int)((nelem & 0xFFFFu) << 16);                   // tile_dim0
    g1[4] = 1;                                                // tile_dim1 = 1
    g1[5] = (int)nelem;                                       // dim0_stride lo32
    g1[6] = 0;
    g1[7] = 0;
    const i32x4 z4 = {0, 0, 0, 0};
#if __clang_major__ >= 23
    const i32x8 z8 = {0, 0, 0, 0, 0, 0, 0, 0};
    __builtin_amdgcn_tensor_load_to_lds(g0, g1, z4, z4, z8, 0);
#else
    __builtin_amdgcn_tensor_load_to_lds(g0, g1, z4, z4, 0);
#endif
}

// ---------------------------------------------------------------------------
// f32 -> bf16 conversion (weights; run once per launch).
// ---------------------------------------------------------------------------
__global__ __launch_bounds__(256)
void cvt_f32_bf16(const float* __restrict__ src, __bf16* __restrict__ dst, int n)
{
    const int i = blockIdx.x * 256 + threadIdx.x;
    if (i < n) dst[i] = (__bf16)src[i];
}

// ---------------------------------------------------------------------------
// Pack NCHW input x (B,96,64,64) f32 -> (pixel, channel) bf16, done once.
// ---------------------------------------------------------------------------
__global__ __launch_bounds__(HID)
void pack_x_bf16(const float* __restrict__ x, __bf16* __restrict__ xp)
{
    const int c   = threadIdx.x;                 // 0..95
    const int pix = blockIdx.x;                  // 0..NBL-1
    const int b   = pix >> 12;
    const int hw  = pix & (LPIX - 1);
    xp[(size_t)pix * HID + c] = (__bf16)x[((size_t)b * HID + c) * LPIX + hw];
}

// ---------------------------------------------------------------------------
// WMMA GEMM:  O[m,n] = sum_k X[m,k] * Wt[n,k];  X,Wt bf16 row-major.
// MODE 1: split halves: col <  N/2 -> O0[m,col], else O1[m,col-N/2]  (f32)
// MODE 2: O0[m,n] = acc * silu(O1[m,n])   (bf16 out if OBF16)
// MODE 3: NCHW store: O0[b,n,hw], m = b*4096+hw  (f32)
// One wave = one 16x16 tile; 8 waves/block share a 16-row A stripe. A frags
// for the full K extent are preloaded per wave. Wave index is readfirstlane'd
// so the n-tile loop and MODE-1 branch are scalar (no EXEC juggling).
// All tiles full (M,N mult of 16) -> EXEC all-ones around every WMMA.
// ---------------------------------------------------------------------------
template <int MODE, int KK, int OBF16>
__global__ __launch_bounds__(256)
void gemm_wmma(const __bf16* __restrict__ X, const __bf16* __restrict__ Wt,
               void* __restrict__ O0v, float* __restrict__ O1, int N)
{
    constexpr int NKT = KK / 32;
    const int lane = threadIdx.x & 31;
    const int wave = __builtin_amdgcn_readfirstlane((int)(threadIdx.x >> 5));
    const int half = lane >> 4;
    const int l15  = lane & 15;
    const int m0   = blockIdx.x * 16;

    // --- preload A fragments for the full K extent (ISA 16-bit A layout) ---
    const __bf16* arow = X + (size_t)(m0 + l15) * KK;
    v16bf afrag[NKT];
#pragma unroll
    for (int kt = 0; kt < NKT; ++kt) {
        const int kb = kt * 32 + half * 8;
        const v8bf lo = *(const v8bf*)(arow + kb);
        const v8bf hi = *(const v8bf*)(arow + kb + 16);
        v16bf a;
#pragma unroll
        for (int i = 0; i < 8; ++i) { a[i] = lo[i]; a[8 + i] = hi[i]; }
        afrag[kt] = a;
    }

    const int ntiles = N >> 4;
    for (int nt = wave; nt < ntiles; nt += 8) {
        const int n0  = nt * 16;
        const int col = n0 + l15;
        const __bf16* wrow = Wt + (size_t)col * KK;
        __builtin_prefetch(wrow, 0, 1);                 // global_prefetch_b8

        v8f acc = {};
#pragma unroll
        for (int kt = 0; kt < NKT; ++kt) {
            const int kb = kt * 32 + half * 8;          // ISA 16-bit B layout
            const v8bf lo = *(const v8bf*)(wrow + kb);
            const v8bf hi = *(const v8bf*)(wrow + kb + 16);
            v16bf b;
#pragma unroll
            for (int i = 0; i < 8; ++i) { b[i] = lo[i]; b[8 + i] = hi[i]; }
            acc = __builtin_amdgcn_wmma_f32_16x16x32_bf16(
                false, afrag[kt], false, b, (short)0, acc, false, false);
        }

        // --- epilogue: lane holds D rows (half*8 + r), column `col` ---
        if (MODE == 1) {
            const int Nh = N >> 1;
            float* O0 = (float*)O0v;
            if (n0 < Nh) {                              // scalar branch (nt SGPR)
#pragma unroll
                for (int r = 0; r < 8; ++r)
                    O0[(size_t)(m0 + half * 8 + r) * Nh + col] = acc[r];
            } else {
#pragma unroll
                for (int r = 0; r < 8; ++r)
                    O1[(size_t)(m0 + half * 8 + r) * Nh + (col - Nh)] = acc[r];
            }
        } else if (MODE == 2) {
#pragma unroll
            for (int r = 0; r < 8; ++r) {
                const int mi = m0 + half * 8 + r;
                const float g = O1[(size_t)mi * N + col];
                const float v = acc[r] * silu_f(g);
                if (OBF16) ((__bf16*)O0v)[(size_t)mi * N + col] = (__bf16)v;
                else       ((float*)O0v)[(size_t)mi * N + col]  = v;
            }
        } else { // MODE 3
            float* O0 = (float*)O0v;
#pragma unroll
            for (int r = 0; r < 8; ++r) {
                const int mi = m0 + half * 8 + r;
                O0[((size_t)(mi >> 12) * N + col) * LPIX + (mi & (LPIX - 1))] = acc[r];
            }
        }
    }
}

// ---------------------------------------------------------------------------
// Depthwise 3x3 conv (SAME) + SiLU.  xin: (B,L,DI) f32 -> xact: (B,L,DI) bf16
// ---------------------------------------------------------------------------
__global__ __launch_bounds__(DI)
void conv2d_dw_silu(const float* __restrict__ xin, const float* __restrict__ w,
                    __bf16* __restrict__ xact)
{
    const int d   = threadIdx.x;
    const int pix = blockIdx.x;
    const int b   = pix >> 12;
    const int hw  = pix & (LPIX - 1);
    const int h   = hw >> 6;
    const int wc  = hw & 63;
    const float* wk = w + d * 9;
    float acc = 0.f;
#pragma unroll
    for (int i = 0; i < 3; ++i) {
        const int hh = h + i - 1;
        if (hh < 0 || hh > 63) continue;
#pragma unroll
        for (int j = 0; j < 3; ++j) {
            const int wcc = wc + j - 1;
            if (wcc < 0 || wcc > 63) continue;
            acc += wk[i * 3 + j] * xin[((size_t)b * LPIX + hh * 64 + wcc) * DI + d];
        }
    }
    xact[(size_t)pix * DI + d] = (__bf16)silu_f(acc);
}

// ---------------------------------------------------------------------------
// Causal depthwise conv1d (k=3, left-pad 2) + bias + SiLU along L (per batch).
// ---------------------------------------------------------------------------
__global__ __launch_bounds__(DI)
void conv1d_dw_silu(const float* __restrict__ xm, const float* __restrict__ w,
                    const float* __restrict__ bias, float* __restrict__ xmc)
{
    const int d   = threadIdx.x;
    const int pix = blockIdx.x;
    const int b   = pix >> 12;
    const int l   = pix & (LPIX - 1);
    float acc = bias[d];
#pragma unroll
    for (int j = 0; j < 3; ++j) {
        const int ll = l - 2 + j;
        if (ll < 0) continue;
        acc += w[d * 3 + j] * xm[((size_t)b * LPIX + ll) * DI + d];
    }
    xmc[(size_t)pix * DI + d] = silu_f(acc);
}

// ---------------------------------------------------------------------------
// x_proj (44x192) + dt_proj (192x12) + softplus per pixel.
// ---------------------------------------------------------------------------
__global__ __launch_bounds__(DI)
void xproj_dt(const float* __restrict__ xmc, const float* __restrict__ xp_w,
              const float* __restrict__ dt_w, const float* __restrict__ dt_b,
              float* __restrict__ DT, float* __restrict__ BM, float* __restrict__ CM)
{
    __shared__ float xrow[DI];
    __shared__ float dbl[DTRANK + 2 * DSTATE];
    const int t = threadIdx.x;
    const size_t pix = blockIdx.x;

    xrow[t] = xmc[pix * DI + t];
    __syncthreads();

    if (t < DTRANK + 2 * DSTATE) {
        const float* wr = xp_w + (size_t)t * DI;
        float acc = 0.f;
#pragma unroll 8
        for (int k = 0; k < DI; ++k) acc += wr[k] * xrow[k];
        dbl[t] = acc;
    }
    __syncthreads();

    float acc = dt_b[t];
#pragma unroll
    for (int r = 0; r < DTRANK; ++r) acc += dt_w[t * DTRANK + r] * dbl[r];
    DT[pix * DI + t] = (acc > 20.f) ? acc : log1pf(__expf(acc));   // softplus

    if (t < DSTATE)          BM[pix * DSTATE + t]            = dbl[DTRANK + t];
    else if (t < 2 * DSTATE) CM[pix * DSTATE + (t - DSTATE)] = dbl[DTRANK + t];
}

// ---------------------------------------------------------------------------
// Selective scan over L=4096. One block per batch, one thread per channel d.
// h[16]/A[16] in registers. B_t/C_t chunks staged into LDS by the Tensor Data
// Mover (wave 0 issues, 1 chunk ahead, double-buffered), synchronized with
// s_wait_tensorcnt + workgroup barrier. Fuses +x*D and *silu(z); bf16 out.
// ---------------------------------------------------------------------------
#define CHUNK 16
__global__ __launch_bounds__(DI)
void scan_kernel(const float* __restrict__ XMC, const float* __restrict__ DT,
                 const float* __restrict__ BM, const float* __restrict__ CM,
                 const float* __restrict__ Z, const float* __restrict__ A_log,
                 const float* __restrict__ Dp, __bf16* __restrict__ Y)
{
    const int d = threadIdx.x;
    const int b = blockIdx.x;
    float A[DSTATE], h[DSTATE];
#pragma unroll
    for (int n = 0; n < DSTATE; ++n) {
        A[n] = -__expf(A_log[d * DSTATE + n]);
        h[n] = 0.f;
    }
    const float Dd = Dp[d];
    __shared__ float sB[2][CHUNK * DSTATE];
    __shared__ float sC[2][CHUNK * DSTATE];
    const size_t base = (size_t)b * LPIX;
    const bool issuer = (threadIdx.x < 32);     // wave 0 drives the TDM
    constexpr int NCH = LPIX / CHUNK;

    if (issuer) {
        tdm_load_f32((unsigned)(uintptr_t)&sB[0][0], BM + base * DSTATE, CHUNK * DSTATE);
        tdm_load_f32((unsigned)(uintptr_t)&sC[0][0], CM + base * DSTATE, CHUNK * DSTATE);
    }
    for (int c = 0; c < NCH; ++c) {
        if (c + 1 < NCH) {
            if (issuer) {
                const int nb = (c + 1) & 1;
                const size_t off = (base + (size_t)(c + 1) * CHUNK) * DSTATE;
                tdm_load_f32((unsigned)(uintptr_t)&sB[nb][0], BM + off, CHUNK * DSTATE);
                tdm_load_f32((unsigned)(uintptr_t)&sC[nb][0], CM + off, CHUNK * DSTATE);
            }
            __builtin_amdgcn_s_wait_tensorcnt(2);   // chunk c's pair retired
        } else {
            __builtin_amdgcn_s_wait_tensorcnt(0);
        }
        __syncthreads();                            // TDM data visible to all waves

        const float* pB = sB[c & 1];
        const float* pC = sC[c & 1];
#pragma unroll 1
        for (int s = 0; s < CHUNK; ++s) {
            const size_t idx = (base + (size_t)c * CHUNK + s) * DI + d;
            const float dt  = DT[idx];
            const float x   = XMC[idx];
            const float dtx = dt * x;
            float y = 0.f;
#pragma unroll
            for (int n = 0; n < DSTATE; ++n) {
                const float da = __expf(dt * A[n]);
                h[n] = da * h[n] + dtx * pB[s * DSTATE + n];
                y += h[n] * pC[s * DSTATE + n];
            }
            const float zz = Z[idx];
            Y[idx] = (__bf16)((y + x * Dd) * silu_f(zz));
        }
        __syncthreads();                            // done reading before reuse
    }
}

// ---------------------------------------------------------------------------
extern "C" void kernel_launch(void* const* d_in, const int* in_sizes, int n_in,
                              void* d_out, int out_size, void* d_ws, size_t ws_size,
                              hipStream_t stream) {
    (void)in_sizes; (void)n_in; (void)out_size; (void)ws_size;

    const float* x      = (const float*)d_in[0];   // (16, 96, 64, 64)
    const float* in_w   = (const float*)d_in[1];   // (384, 96)
    const float* c2w    = (const float*)d_in[2];   // (192, 1, 3, 3)
    const float* min_w  = (const float*)d_in[3];   // (384, 192)
    const float* c1w    = (const float*)d_in[4];   // (192, 1, 3)
    const float* c1b    = (const float*)d_in[5];   // (192,)
    const float* xp_w   = (const float*)d_in[6];   // (44, 192)
    const float* dtw    = (const float*)d_in[7];   // (192, 12)
    const float* dtb    = (const float*)d_in[8];   // (192,)
    const float* Alog   = (const float*)d_in[9];   // (192, 16)
    const float* Dp     = (const float*)d_in[10];  // (192,)
    const float* mout_w = (const float*)d_in[11];  // (192, 192)
    const float* out_w  = (const float*)d_in[12];  // (96, 192)
    float* out = (float*)d_out;

    // Workspace layout (slot = SZ floats) with lifetime-based reuse.
    float* ws   = (float*)d_ws;
    const size_t SZ = (size_t)NBL * DI;
    float*  XIN  = ws;                     // slot0: in_proj half 0 (f32)
    float*  GATE = ws + 1 * SZ;            // slot1: x_gate (f32, lives to GEMM7)
    __bf16* XACT = (__bf16*)(ws + 2 * SZ); // slot2: conv2d+silu out (bf16)
    float*  XM   = ws + 3 * SZ;            // slot3: m_in_proj half 0 (f32)
    float*  Z    = ws + 4 * SZ;            // slot4: m_in_proj half 1 (f32)
    float*  BM   = ws + 5 * SZ;            // (NBL, 16) f32
    float*  CM   = BM + (size_t)NBL * DSTATE;
    float*  XMC  = XIN;                    // slot0 reuse (XIN dead after conv2d)
    float*  DT   = (float*)(ws + 2 * SZ);  // slot2 reuse (XACT dead after GEMM3)
    __bf16* Y    = (__bf16*)XM;            // slot3 reuse (XM dead after conv1d)
    __bf16* YM   = (__bf16*)Z;             // slot4 reuse (Z dead after scan)

    // bf16 staging: converted weights + packed input, after BM/CM.
    __bf16* WB_IN  = (__bf16*)(CM + (size_t)NBL * DSTATE);
    __bf16* WB_MIN = WB_IN  + (size_t)DI2 * HID;   // 384*96
    __bf16* WB_MOUT= WB_MIN + (size_t)DI2 * DI;    // 384*192
    __bf16* WB_OUT = WB_MOUT+ (size_t)DI  * DI;    // 192*192
    __bf16* XP     = WB_OUT + (size_t)HID * DI;    // 96*192 -> then NBL*96

    const dim3 gGemm(NBL / 16), bGemm(256);
    const dim3 gPix(NBL), bPix(DI);

    // 0) one-time per launch: weights -> bf16, x -> (pixel, channel) bf16
    cvt_f32_bf16<<<(DI2 * HID + 255) / 256, 256, 0, stream>>>(in_w,   WB_IN,   DI2 * HID);
    cvt_f32_bf16<<<(DI2 * DI  + 255) / 256, 256, 0, stream>>>(min_w,  WB_MIN,  DI2 * DI);
    cvt_f32_bf16<<<(DI  * DI  + 255) / 256, 256, 0, stream>>>(mout_w, WB_MOUT, DI  * DI);
    cvt_f32_bf16<<<(HID * DI  + 255) / 256, 256, 0, stream>>>(out_w,  WB_OUT,  HID * DI);
    pack_x_bf16<<<gPix, dim3(HID), 0, stream>>>(x, XP);

    // 1) in_proj: XP @ WB_IN^T -> XIN | GATE                  (WMMA)
    gemm_wmma<1, HID, 0><<<gGemm, bGemm, 0, stream>>>(XP, WB_IN, XIN, GATE, DI2);
    // 2) depthwise 3x3 conv + SiLU -> bf16
    conv2d_dw_silu<<<gPix, bPix, 0, stream>>>(XIN, c2w, XACT);
    // 3) m_in_proj: XACT @ WB_MIN^T -> XM | Z                 (WMMA)
    gemm_wmma<1, DI, 0><<<gGemm, bGemm, 0, stream>>>(XACT, WB_MIN, XM, Z, DI2);
    // 4) causal conv1d + bias + SiLU
    conv1d_dw_silu<<<gPix, bPix, 0, stream>>>(XM, c1w, c1b, XMC);
    // 5) x_proj + dt_proj + softplus -> DT, BM, CM
    xproj_dt<<<gPix, bPix, 0, stream>>>(XMC, xp_w, dtw, dtb, DT, BM, CM);
    // 6) selective scan (TDM-staged B/C, fused epilogues) -> Y (bf16)
    scan_kernel<<<dim3(B_SZ), bPix, 0, stream>>>(XMC, DT, BM, CM, Z, Alog, Dp, Y);
    // 7) m_out_proj with outer gate: YM = (Y @ WB_MOUT^T)*silu(GATE)  (WMMA, bf16)
    gemm_wmma<2, DI, 1><<<gGemm, bGemm, 0, stream>>>(Y, WB_MOUT, YM, GATE, DI);
    // 8) out_proj + NCHW transpose store                              (WMMA)
    gemm_wmma<3, DI, 0><<<gGemm, bGemm, 0, stream>>>(YM, WB_OUT, out, nullptr, HID);
}
